// ContTimeLSTM_27590869910090
// MI455X (gfx1250) — matrix operand order repaired and checked by reading, not compiled, not run on hardware
//
#include <hip/hip_runtime.h>

// ---------------------------------------------------------------------------
// Continuous-time LSTM scan for MI455X (gfx1250, wave32, WMMA bf16).
// B=128, L=256, D=512, S=512, K=1024.
//
// 32 persistent "owner" workgroups, each owning 16 state columns and their 6
// gate slices of the 3072-wide projection. The bf16 weight slice (96 rows x
// 1024 K = 198KB) is parked in LDS once and reused for all 256 steps.
// ONE atomic grid barrier per step (h broadcast); the d-scalar GEMV is
// computed redundantly per-WG (~1% extra FLOPs) to avoid a second barrier.
// x is pre-converted to bf16 [L][B][D] in a prepass when workspace allows.
// ---------------------------------------------------------------------------

#define B_   128
#define L_   256
#define D_   512
#define S_   512
#define KTOT 1024
#define NWG  32          // 32 owner workgroups (column tiles of 16)
#define NTH  512         // 16 waves of 32
#define WSTR 2064        // LDS bytes per weight row (2048 + 16 pad)
#define FS0  ((size_t)B_ * L_ * S_)   // final_state offset in d_out

typedef __attribute__((ext_vector_type(16))) __bf16       bf16x16;
typedef __attribute__((ext_vector_type(4)))  __bf16       bf16x4;
typedef __attribute__((ext_vector_type(8)))  float        f32x8;
typedef __attribute__((ext_vector_type(4)))  float        f32x4;
typedef __attribute__((ext_vector_type(4)))  unsigned int u32x4;

union Frag  { u32x4 u[2]; bf16x16 v; __bf16 h[16]; };
union Pack8 { u32x4 u;    __bf16 h[8]; };

// -------- shared memory layout (bytes) --------
#define OFF_W    0                        // weights: 96 * WSTR   = 198144
#define OFF_P    (96 * WSTR)              // proj 6*2048 f32      = 49152
#define OFF_C    (OFF_P  + 6 * 2048 * 4)  // c     2048 f32       = 8192
#define OFF_O    (OFF_C  + 2048 * 4)
#define OFF_CS   (OFF_O  + 2048 * 4)
#define OFF_CE   (OFF_CS + 2048 * 4)
#define OFF_DW   (OFF_CE + 2048 * 4)      // dw 1024 f32          = 4096
#define OFF_DD   (OFF_DW + 1024 * 4)      // shD 128 f32          = 512
#define OFF_RED  (OFF_DD + 128 * 4)       // shRed 512 f32        = 2048
#define SMEM_SZ  (OFF_RED + 512 * 4)      // = 286720 < 320KB

extern __shared__ char smem[];

__device__ __forceinline__ float sigf(float v) {
    return 1.0f / (1.0f + __expf(-v));
}

// Monotonic grid barrier (one counter per step, zeroed each launch).
__device__ __forceinline__ void grid_barrier(unsigned int* cnt, int idx) {
    __threadfence();
    __syncthreads();
    if (threadIdx.x == 0) {
        __hip_atomic_fetch_add(&cnt[idx], 1u, __ATOMIC_ACQ_REL,
                               __HIP_MEMORY_SCOPE_AGENT);
        while (__hip_atomic_load(&cnt[idx], __ATOMIC_ACQUIRE,
                                 __HIP_MEMORY_SCOPE_AGENT) < NWG) {
            __builtin_amdgcn_s_sleep(1);
        }
    }
    __syncthreads();
}

__global__ void ctlstm_init(unsigned int* cnt) {
    int i = threadIdx.x + blockIdx.x * blockDim.x;
    if (i < L_) cnt[i] = 0u;
}

// prepass: x [B,L,D] fp32 -> xb [L,B,D] bf16
__global__ void ctlstm_xcvt(const float* __restrict__ x,
                            __bf16* __restrict__ xb) {
    size_t i = (size_t)blockIdx.x * blockDim.x + threadIdx.x;   // over n/4
    size_t n4 = (size_t)B_ * L_ * D_ / 4;
    if (i >= n4) return;
    size_t k4 = i % (D_ / 4);
    size_t bt = i / (D_ / 4);
    size_t b  = bt / L_;
    size_t t  = bt % L_;
    f32x4 v = *((const f32x4*)x + i);
    bf16x4 o;
    o[0] = (__bf16)v[0]; o[1] = (__bf16)v[1];
    o[2] = (__bf16)v[2]; o[3] = (__bf16)v[3];
    *(bf16x4*)(xb + (t * B_ + b) * D_ + k4 * 4) = o;
}

__global__ __launch_bounds__(NTH, 1)
void ctlstm_kernel(const float* __restrict__ x,      // [B,L,D]
                   const float* __restrict__ td,     // [B,L]
                   const float* __restrict__ w,      // [6S+1, K]
                   const float* __restrict__ bias,   // [6S+1]
                   const float* __restrict__ dw,     // [K]
                   const float* __restrict__ dbias,  // [1]
                   const float* __restrict__ dbeta,  // scalar
                   float*       __restrict__ out,    // outputs ++ final_state
                   __bf16*      __restrict__ hb,     // [2][B][S] bf16 (ws)
                   const __bf16* __restrict__ xb,    // [L][B][D] bf16 or null
                   unsigned int* __restrict__ cnt)   // [L] (ws)
{
    const int jt  = blockIdx.x;
    const int tid = threadIdx.x;

    float* shP  = (float*)(smem + OFF_P);
    float* shC  = (float*)(smem + OFF_C);
    float* shO  = (float*)(smem + OFF_O);
    float* shCS = (float*)(smem + OFF_CS);
    float* shCE = (float*)(smem + OFF_CE);
    float* shDW = (float*)(smem + OFF_DW);
    float* shD  = (float*)(smem + OFF_DD);
    float* shRd = (float*)(smem + OFF_RED);

    // ---- one-time init: zero state, preload weights (bf16) + dw into LDS ---
    for (int i = tid; i < 2048; i += NTH) {
        shO[i] = 0.0f; shCS[i] = 0.0f; shCE[i] = 0.0f;
    }
    for (int i = tid; i < 1024; i += NTH) shDW[i] = dw[i];
    if (tid < 128) shD[tid] = 0.0f;
    for (int i = tid; i < 96 * 512; i += NTH) {
        int rl = i >> 9;            // local row 0..95 (gate*16 + col)
        int kp = (i & 511) * 2;     // K pair
        int g  = rl >> 4;
        int n  = g * S_ + jt * 16 + (rl & 15);
        const float* wr = w + (size_t)n * KTOT + kp;
        __bf16* dst = (__bf16*)(smem + OFF_W + (size_t)rl * WSTR + (size_t)kp * 2);
        dst[0] = (__bf16)wr[0];
        dst[1] = (__bf16)wr[1];
    }

    const int lane = tid & 31;
    const int wv   = tid >> 5;      // wave 0..15
    const int mt   = wv >> 1;       // batch tile 0..7
    const int g2   = wv & 1;        // gates 0..2 or 3..5
    const int nlo  = lane & 15;
    const int half = lane >> 4;
    const int mrow = mt * 16 + nlo;

    const float bs0 = bias[(g2 * 3 + 0) * S_ + jt * 16 + nlo];
    const float bs1 = bias[(g2 * 3 + 1) * S_ + jt * 16 + nlo];
    const float bs2 = bias[(g2 * 3 + 2) * S_ + jt * 16 + nlo];
    const float beta = *dbeta;
    const float db0  = *dbias;

    const int jl   = tid & 15;      // elementwise mapping
    const int brow = tid >> 4;      // 0..31
    const int jg   = jt * 16 + jl;
    const int q    = tid & 3;       // d-GEMV mapping: 4 threads per batch
    const int bq   = tid >> 2;

    const char* wb0base = smem + OFF_W + (size_t)((g2 * 3) * 16 + nlo) * WSTR;

    __syncthreads();

    for (int t = 0; t < L_; ++t) {
        const int par = t & 1;

        // ---- phase A: c, h from carry; publish h ----
        #pragma unroll
        for (int e = 0; e < 4; ++e) {
            int b   = brow + e * 32;
            int idx = b * 16 + jl;
            float dno = shD[b];
            float dtv = td[b * L_ + t];
            float cs = shCS[idx], ce = shCE[idx];
            float c  = cs + (ce - cs) * __expf(-dno * dtv);
            shC[idx] = c;
            float hv = shO[idx] * (2.0f * sigf(2.0f * c) - 1.0f);
            out[(size_t)b * (L_ * S_) + (size_t)t * S_ + jg] = hv;
            hb[(size_t)par * (B_ * S_) + (size_t)b * S_ + jg] = (__bf16)hv;
        }
        grid_barrier(cnt, t);   // h visible device-wide

        // ---- phase B: GEMM over K=1024 (branch-free split loops) ----
        f32x8 a0 = {}, a1 = {}, a2 = {};
        const __bf16* hrow = hb + (size_t)par * (B_ * S_) + (size_t)mrow * S_;

        if (xb) {
            const __bf16* xrow = xb + ((size_t)t * B_ + mrow) * D_;
            #pragma unroll 4
            for (int kk = 0; kk < 16; ++kk) {
                const int kc = kk * 32;
                Frag a;
                const __bf16* p = xrow + kc + half * 8;
                a.u[0] = *(const u32x4*)(p);
                a.u[1] = *(const u32x4*)(p + 16);
                const char* pb = wb0base + (size_t)(kc + half * 8) * 2;
                Frag w0, w1, w2;
                w0.u[0] = *(const u32x4*)(pb);
                w0.u[1] = *(const u32x4*)(pb + 32);
                w1.u[0] = *(const u32x4*)(pb + 16 * WSTR);
                w1.u[1] = *(const u32x4*)(pb + 16 * WSTR + 32);
                w2.u[0] = *(const u32x4*)(pb + 32 * WSTR);
                w2.u[1] = *(const u32x4*)(pb + 32 * WSTR + 32);
                a0 = __builtin_amdgcn_wmma_f32_16x16x32_bf16(
                        false, a.v, false, w0.v, (short)0, a0, false, false);
                a1 = __builtin_amdgcn_wmma_f32_16x16x32_bf16(
                        false, a.v, false, w1.v, (short)0, a1, false, false);
                a2 = __builtin_amdgcn_wmma_f32_16x16x32_bf16(
                        false, a.v, false, w2.v, (short)0, a2, false, false);
            }
            // prefetch next step's x slab into caches during compute
            if (t + 1 < L_) {
                const char* nx = (const char*)(xb + ((size_t)(t + 1) * B_ + mrow) * D_);
                __builtin_prefetch(nx + (size_t)lane * 64, 0, 1);
            }
        } else {
            const float* xrow = x + ((size_t)mrow * L_ + t) * D_;
            #pragma unroll 4
            for (int kk = 0; kk < 16; ++kk) {
                const int kc = kk * 32;
                Frag a;
                const float* p = xrow + kc + half * 8;
                f32x4 f0 = *(const f32x4*)(p);
                f32x4 f1 = *(const f32x4*)(p + 4);
                f32x4 f2 = *(const f32x4*)(p + 16);
                f32x4 f3 = *(const f32x4*)(p + 20);
                #pragma unroll
                for (int i = 0; i < 4; ++i) {
                    a.h[i]      = (__bf16)f0[i];
                    a.h[4 + i]  = (__bf16)f1[i];
                    a.h[8 + i]  = (__bf16)f2[i];
                    a.h[12 + i] = (__bf16)f3[i];
                }
                const char* pb = wb0base + (size_t)(kc + half * 8) * 2;
                Frag w0, w1, w2;
                w0.u[0] = *(const u32x4*)(pb);
                w0.u[1] = *(const u32x4*)(pb + 32);
                w1.u[0] = *(const u32x4*)(pb + 16 * WSTR);
                w1.u[1] = *(const u32x4*)(pb + 16 * WSTR + 32);
                w2.u[0] = *(const u32x4*)(pb + 32 * WSTR);
                w2.u[1] = *(const u32x4*)(pb + 32 * WSTR + 32);
                a0 = __builtin_amdgcn_wmma_f32_16x16x32_bf16(
                        false, a.v, false, w0.v, (short)0, a0, false, false);
                a1 = __builtin_amdgcn_wmma_f32_16x16x32_bf16(
                        false, a.v, false, w1.v, (short)0, a1, false, false);
                a2 = __builtin_amdgcn_wmma_f32_16x16x32_bf16(
                        false, a.v, false, w2.v, (short)0, a2, false, false);
            }
        }
        #pragma unroll 4
        for (int kk = 16; kk < 32; ++kk) {
            const int kc = kk * 32;
            Frag a;
            const __bf16* p = hrow + (kc - D_) + half * 8;
            a.u[0] = *(const u32x4*)(p);
            a.u[1] = *(const u32x4*)(p + 16);
            const char* pb = wb0base + (size_t)(kc + half * 8) * 2;
            Frag w0, w1, w2;
            w0.u[0] = *(const u32x4*)(pb);
            w0.u[1] = *(const u32x4*)(pb + 32);
            w1.u[0] = *(const u32x4*)(pb + 16 * WSTR);
            w1.u[1] = *(const u32x4*)(pb + 16 * WSTR + 32);
            w2.u[0] = *(const u32x4*)(pb + 32 * WSTR);
            w2.u[1] = *(const u32x4*)(pb + 32 * WSTR + 32);
            a0 = __builtin_amdgcn_wmma_f32_16x16x32_bf16(
                    false, a.v, false, w0.v, (short)0, a0, false, false);
            a1 = __builtin_amdgcn_wmma_f32_16x16x32_bf16(
                    false, a.v, false, w1.v, (short)0, a1, false, false);
            a2 = __builtin_amdgcn_wmma_f32_16x16x32_bf16(
                    false, a.v, false, w2.v, (short)0, a2, false, false);
        }

        // bias + sigmoid, stage gates to LDS (D layout: VGPR r -> M = r+8*half)
        #pragma unroll
        for (int r = 0; r < 8; ++r) {
            int b   = mt * 16 + half * 8 + r;
            int idx = b * 16 + nlo;
            shP[(g2 * 3 + 0) * 2048 + idx] = sigf(a0[r] + bs0);
            shP[(g2 * 3 + 1) * 2048 + idx] = sigf(a1[r] + bs1);
            shP[(g2 * 3 + 2) * 2048 + idx] = sigf(a2[r] + bs2);
        }

        // ---- redundant in-WG d-GEMV partials (avoids 2nd grid barrier) ----
        {
            float part = 0.0f;
            if (xb) {
                const __bf16* xq = xb + ((size_t)t * B_ + bq) * D_ + q * 128;
                #pragma unroll
                for (int c = 0; c < 16; ++c) {
                    Pack8 v; v.u = *(const u32x4*)(xq + c * 8);
                    #pragma unroll
                    for (int j = 0; j < 8; ++j)
                        part += (float)v.h[j] * shDW[q * 128 + c * 8 + j];
                }
            } else {
                const float* xq = x + ((size_t)bq * L_ + t) * D_ + q * 128;
                #pragma unroll
                for (int c = 0; c < 32; ++c) {
                    f32x4 v = *(const f32x4*)(xq + c * 4);
                    #pragma unroll
                    for (int j = 0; j < 4; ++j)
                        part += v[j] * shDW[q * 128 + c * 4 + j];
                }
            }
            const __bf16* hq = hb + (size_t)par * (B_ * S_)
                                  + (size_t)bq * S_ + q * 128;
            #pragma unroll
            for (int c = 0; c < 16; ++c) {
                Pack8 v; v.u = *(const u32x4*)(hq + c * 8);
                #pragma unroll
                for (int j = 0; j < 8; ++j)
                    part += (float)v.h[j] * shDW[D_ + q * 128 + c * 8 + j];
            }
            shRd[tid] = part;
        }
        __syncthreads();   // proj + shRd visible

        // ---- phase C: state update ----
        #pragma unroll
        for (int e = 0; e < 4; ++e) {
            int b   = brow + e * 32;
            int idx = b * 16 + jl;
            float c   = shC[idx];
            float ig  = shP[idx];
            float fg  = shP[2048 + idx];
            float ieg = shP[2 * 2048 + idx];
            float feg = shP[3 * 2048 + idx];
            float z   = 2.0f * shP[4 * 2048 + idx] - 1.0f;
            float on  = shP[5 * 2048 + idx];
            shCS[idx] = fg * c + ig * z;
            shCE[idx] = feg * shCE[idx] + ieg * z;
            shO[idx]  = on;
        }
        // ---- finalize d (softplus) ----
        if (tid < 128) {
            float v  = shRd[4 * tid] + shRd[4 * tid + 1]
                     + shRd[4 * tid + 2] + shRd[4 * tid + 3] + db0;
            float aa = beta * v;
            float sp = fmaxf(aa, 0.0f) + __logf(1.0f + __expf(-fabsf(aa)));
            shD[tid] = sp / beta;
        }
        __syncthreads();   // state + d ready for next phase A
    }

    // ---- final state: o | cs | ce (+ d from WG 0) ----
    #pragma unroll
    for (int e = 0; e < 4; ++e) {
        int b   = brow + e * 32;
        int idx = b * 16 + jl;
        size_t base = FS0 + (size_t)b * (3 * S_ + 1);
        out[base + jg]          = shO[idx];
        out[base + S_ + jg]     = shCS[idx];
        out[base + 2 * S_ + jg] = shCE[idx];
    }
    if (jt == 0 && tid < 128)
        out[FS0 + (size_t)tid * (3 * S_ + 1) + 3 * S_] = shD[tid];
}

extern "C" void kernel_launch(void* const* d_in, const int* in_sizes, int n_in,
                              void* d_out, int out_size, void* d_ws, size_t ws_size,
                              hipStream_t stream) {
    const float* x     = (const float*)d_in[0];
    const float* td    = (const float*)d_in[1];
    const float* w     = (const float*)d_in[2];
    const float* bias  = (const float*)d_in[3];
    const float* dw    = (const float*)d_in[4];
    const float* dbias = (const float*)d_in[5];
    const float* dbeta = (const float*)d_in[6];
    float* out = (float*)d_out;

    // workspace: hb (2*B*S bf16 = 256KB) | cnt (1KB) | [optional xb 33.5MB]
    char* ws = (char*)d_ws;
    __bf16*       hb  = (__bf16*)ws;
    unsigned int* cnt = (unsigned int*)(ws + 2 * B_ * S_ * 2);
    size_t fixed = 2 * (size_t)B_ * S_ * 2 + 4096;
    size_t xbytes = (size_t)L_ * B_ * D_ * 2;
    __bf16* xbf = nullptr;
    if (ws_size >= fixed + xbytes) xbf = (__bf16*)(ws + fixed);

    ctlstm_init<<<1, 256, 0, stream>>>(cnt);
    if (xbf) {
        size_t n4 = (size_t)B_ * L_ * D_ / 4;
        ctlstm_xcvt<<<(int)((n4 + 255) / 256), 256, 0, stream>>>(x, xbf);
    }
    ctlstm_kernel<<<NWG, NTH, SMEM_SZ, stream>>>(
        x, td, w, bias, dw, dbias, dbeta, out, hb, xbf, cnt);
}
